// LocSE_57732950393239
// MI455X (gfx1250) — matrix operand order repaired and checked by reading, not compiled, not run on hardware
//
#include <hip/hip_runtime.h>

typedef _Float16 v16h __attribute__((ext_vector_type(16)));
typedef _Float16 v8h  __attribute__((ext_vector_type(8)));
typedef float    v8f  __attribute__((ext_vector_type(8)));
typedef float    v4f  __attribute__((ext_vector_type(4)));
typedef int      v4i  __attribute__((ext_vector_type(4)));

// ---- workspace layout (in f16 elements) ----
#define WS_NRM0 0          // 64*32
#define WS_NRM1 2048       // 64*192
#define WS_NRM2 14336      // 64*192
#define WS_ANG0 26624      // 64*32
#define WS_ANG1 28672      // 64*192
#define WS_ANG2 40960      // 64*192
#define WS_MLP  53248      // 64*128
// total 61440 halves = 122880 bytes

#if __has_builtin(__builtin_amdgcn_global_load_async_to_lds_b128) && \
    __has_builtin(__builtin_amdgcn_s_wait_asynccnt)
#define USE_ASYNC_LDS 1
#else
#define USE_ASYNC_LDS 0
#endif

__device__ __forceinline__ v8f wmma16(v16h a, v16h b, v8f c) {
  return __builtin_amdgcn_wmma_f32_16x16x32_f16(false, a, false, b, (short)0, c,
                                                false, false);
}

// -------- weight prep: fold column-sum (x + sum(x) trick), transpose, f16 --------
// conv (cin>0): w is (3,cin,64) row-major; dst[o][K], K=t*cin+c, padded to kpad
// mlp (cin==0): w is (kpad,64); dst[o][K] = w[K][o]
__global__ void prep_weights(const float* __restrict__ w, _Float16* __restrict__ dst,
                             int cin, int kpad) {
  int idx = blockIdx.x * blockDim.x + threadIdx.x;
  if (idx >= 64 * kpad) return;
  int o = idx / kpad;
  int K = idx % kpad;
  float v = 0.f;
  if (cin > 0) {
    if (K < 3 * cin) {
      int t = K / cin, c = K % cin;
      float colsum = 0.f;
      for (int cc = 0; cc < cin; ++cc) colsum += w[(t * cin + cc) * 64 + o];
      v = w[(t * cin + c) * 64 + o] + colsum;
    }
  } else {
    v = w[K * 64 + o];
  }
  dst[o * kpad + K] = (_Float16)v;
}

// -------- feats pass-through: out[R*128 + 64 + c] = feats[R*64 + c] --------
// Pure streaming copy: non-temporal both ways so it does not pollute L2.
__global__ void feats_copy(const v4f* __restrict__ feats, float* __restrict__ out) {
  long i = (long)blockIdx.x * 256 + threadIdx.x;   // 11,520,000 float4s (exact)
  long e = i << 2;                                  // element index = R*64 + c
  long R = e >> 6;
  int  c = (int)(e & 63);
  v4f v = __builtin_nontemporal_load(feats + i);
  __builtin_nontemporal_store(v, (v4f*)(out + R * 128 + 64 + c));
}

// -------- weight stage into LDS: async DMA path when available --------
__device__ __forceinline__ void stage_w(const _Float16* __restrict__ src,
                                        _Float16* dstw, int halves, int tid) {
#if USE_ASYNC_LDS
  int n = halves >> 3;                              // 16B chunks
  for (int i = tid; i < n; i += 256) {
    __builtin_amdgcn_global_load_async_to_lds_b128(
        (__attribute__((address_space(1))) v4i*)(src + i * 8),
        (__attribute__((address_space(3))) v4i*)(dstw + i * 8), 0, 0);
  }
#else
  const int4* s = (const int4*)src;
  int4* d = (int4*)dstw;
  int n = halves >> 3;
  for (int i = tid; i < n; i += 256) d[i] = s[i];
#endif
}

__device__ __forceinline__ void stage_fence() {
#if USE_ASYNC_LDS
  __builtin_amdgcn_s_wait_asynccnt(0);
#endif
  __syncthreads();
}

// -------- layer0 conv (tiny cin, gathered A fragment), relu, no residual --------
template <int CIN>
__device__ __forceinline__ void conv0_layer(const _Float16* s_in, const _Float16* s_w,
                                            _Float16* s_out, const float* __restrict__ bias,
                                            int chbase, int tid) {
  const int lane = tid & 31, wave = tid >> 5;
  const int hi = lane >> 4, ln = lane & 15;
  for (int j = wave; j < 36; j += 8) {
    int rt = j % 9, nt = j / 9;
    int r = rt * 16 + ln, p = r / 9, s = r % 9;
    v16h a;
#pragma unroll
    for (int h = 0; h < 16; ++h) {
      int K = h + ((h >= 8) ? 8 : 0) + hi * 8;    // A layout: 16-bit 16x32
      bool valid = (K < 3 * CIN);
      int t = valid ? (K / CIN) : 0;
      int c = valid ? (K % CIN) : 0;
      _Float16 v = s_in[(p * 11 + s + t) * 4 + chbase + c];
      a[h] = valid ? v : (_Float16)0.f;
    }
    v16h b;
    {
      const _Float16* wp = s_w + (nt * 16 + ln) * 32 + hi * 8;  // B: K=8*hi+h (+16 for h>=8), N=ln
      v8h lo = *(const v8h*)wp;
      v8h hh = *(const v8h*)(wp + 16);
#pragma unroll
      for (int i = 0; i < 8; ++i) { b[i] = lo[i]; b[i + 8] = hh[i]; }
    }
    v8f acc = {};
    acc = wmma16(a, b, acc);
    int n = nt * 16 + ln;
    float bn = bias[n];
#pragma unroll
    for (int i = 0; i < 8; ++i) {                 // D: M = i + 8*hi, N = ln
      int rr = rt * 16 + i + hi * 8, pp = rr / 9, ss = rr % 9;
      float val = fmaxf(acc[i] + bn, 0.f);
      s_out[(pp * 11 + ss + 1) * 64 + n] = (_Float16)val;
    }
  }
}

// -------- residual conv layer (cin=64, K=192 = 6 WMMA K-tiles) --------
// HOLD: additionally mirror the result into s_hold[144][64] (used to keep nrm_e)
template <bool HOLD>
__device__ __forceinline__ void convN_layer(const _Float16* s_ain, const _Float16* s_w,
                                            _Float16* s_out, const float* __restrict__ bias,
                                            _Float16* s_hold, int tid) {
  const int lane = tid & 31, wave = tid >> 5;
  const int hi = lane >> 4, ln = lane & 15;
  for (int j = wave; j < 36; j += 8) {
    int rt = j % 9, nt = j / 9;
    int r = rt * 16 + ln, p = r / 9, s = r % 9;
    const _Float16* base = s_ain + (p * 11 + s) * 64;  // tap t -> +t*64 (padded rows)
    int n = nt * 16 + ln;
    const _Float16* wrow = s_w + n * 192;
    v8f acc = {};
#pragma unroll
    for (int kt = 0; kt < 6; ++kt) {
      int K0 = kt * 32 + hi * 8;                  // run of 8 consecutive K
      int K1 = K0 + 16;
      v8h alo = *(const v8h*)(base + (K0 >> 6) * 64 + (K0 & 63));
      v8h ahi = *(const v8h*)(base + (K1 >> 6) * 64 + (K1 & 63));
      v16h a;
#pragma unroll
      for (int i = 0; i < 8; ++i) { a[i] = alo[i]; a[i + 8] = ahi[i]; }
      const _Float16* wp = wrow + kt * 32 + hi * 8;
      v8h blo = *(const v8h*)wp;
      v8h bhi = *(const v8h*)(wp + 16);
      v16h b;
#pragma unroll
      for (int i = 0; i < 8; ++i) { b[i] = blo[i]; b[i + 8] = bhi[i]; }
      acc = wmma16(a, b, acc);
    }
    float bn = bias[n];
#pragma unroll
    for (int i = 0; i < 8; ++i) {
      int rr = rt * 16 + i + hi * 8, pp = rr / 9, ss = rr % 9;
      int idx = (pp * 11 + ss + 1) * 64 + n;
      float prev = (float)s_ain[idx];
      float val = prev + fmaxf(acc[i] + bn, 0.f);
      _Float16 hv = (_Float16)val;
      s_out[idx] = hv;
      if (HOLD) s_hold[rr * 64 + n] = hv;
    }
  }
}

// -------- final MLP: concat(nrm_e, ang_e) [K=128] @ Wt, relu, write f32 out --------
__device__ __forceinline__ void mlp_layer(const _Float16* s_hold, const _Float16* s_ang,
                                          const _Float16* s_w, const float* __restrict__ bias,
                                          float* __restrict__ out, long blockBase, int tid) {
  const int lane = tid & 31, wave = tid >> 5;
  const int hi = lane >> 4, ln = lane & 15;
  for (int j = wave; j < 36; j += 8) {
    int rt = j % 9, nt = j / 9;
    int r = rt * 16 + ln, p = r / 9, s = r % 9;
    int n = nt * 16 + ln;
    const _Float16* wrow = s_w + n * 128;
    v8f acc = {};
#pragma unroll
    for (int kt = 0; kt < 4; ++kt) {
      int K0 = (kt & 1) * 32 + hi * 8;
      const _Float16* abase = (kt < 2) ? (s_hold + r * 64)
                                       : (s_ang + (p * 11 + s + 1) * 64);
      v8h alo = *(const v8h*)(abase + K0);
      v8h ahi = *(const v8h*)(abase + K0 + 16);
      v16h a;
#pragma unroll
      for (int i = 0; i < 8; ++i) { a[i] = alo[i]; a[i + 8] = ahi[i]; }
      const _Float16* wp = wrow + kt * 32 + hi * 8;
      v8h blo = *(const v8h*)wp;
      v8h bhi = *(const v8h*)(wp + 16);
      v16h b;
#pragma unroll
      for (int i = 0; i < 8; ++i) { b[i] = blo[i]; b[i + 8] = bhi[i]; }
      acc = wmma16(a, b, acc);
    }
    float bn = bias[n];
#pragma unroll
    for (int i = 0; i < 8; ++i) {
      long R = blockBase + rt * 16 + i + hi * 8;
      float val = fmaxf(acc[i] + bn, 0.f);
      __builtin_nontemporal_store(val, out + R * 128 + n);  // streamed once
    }
  }
}

// ================== main fused kernel: 16 points (144 rows) per block ==================
// Weight panels double-buffered in LDS; layer i+1 is async-prefetched (ASYNCcnt)
// while layer i's WMMAs run; s_wait_asynccnt 0 + barrier at each phase edge.
__global__ __launch_bounds__(256)
void locse_main(const float* __restrict__ pc, const _Float16* __restrict__ ws,
                const float* __restrict__ nrm_b0, const float* __restrict__ nrm_b1,
                const float* __restrict__ nrm_b2, const float* __restrict__ ang_b0,
                const float* __restrict__ ang_b1, const float* __restrict__ ang_b2,
                const float* __restrict__ mlp_b, float* __restrict__ out) {
  __shared__ __align__(16) _Float16 s_in[16 * 11 * 4];        // padded inputs (4 ch)
  __shared__ __align__(16) _Float16 s_act[2][16 * 11 * 64];   // ping-pong activations
  __shared__ __align__(16) _Float16 s_w[2][64 * 192];         // double-buffered weights
  __shared__ __align__(16) _Float16 s_hold[144 * 64];         // nrm_e result

  const int tid = threadIdx.x;

  // kick off first weight panel while we zero-init and build inputs
  stage_w(ws + WS_NRM0, s_w[0], 2048, tid);

  // zero activation buffers + input buffer (establishes zero S-padding)
  {
    unsigned int* z0 = (unsigned int*)&s_act[0][0];
    for (int i = tid; i < 11264; i += 256) z0[i] = 0u;
    unsigned int* z1 = (unsigned int*)s_in;
    for (int i = tid; i < 352; i += 256) z1[i] = 0u;
  }
  __syncthreads();

  // pointwise geometry features -> s_in[p][s+1][{nrm, ang0, ang1, ang2}]
  if (tid < 144) {
    int p = tid / 9, s = tid % 9;
    long gp = (long)blockIdx.x * 16 + p;
    float4 row = *(const float4*)(pc + (gp * 9 + s) * 4);
    float4 cur = *(const float4*)(pc + (gp * 9) * 4);
    float d0 = cur.x - row.x, d1 = cur.y - row.y;
    float sp = row.x * row.x + row.y * row.y;
    float norms = (sp == 0.f) ? 0.f : sqrtf(sp);
    float sd = d0 * d0 + d1 * d1;
    float dn = (sd == 0.f) ? 0.f : sqrtf(sd);
    float sc = cur.x * cur.x + cur.y * cur.y + cur.z * cur.z + cur.w * cur.w;
    float cn = (sc == 0.f) ? 0.f : sqrtf(sc);
    float num = d0 * cur.z + d1 * cur.w;
    float den = dn * cn + 1e-8f;
    float ang = 1.f - fabsf(num / den);
    _Float16* dst = &s_in[(p * 11 + s + 1) * 4];
    dst[0] = (_Float16)d0;     // nrm_in
    dst[1] = (_Float16)d1;     // ang_in[0]
    dst[2] = (_Float16)norms;  // ang_in[1]
    dst[3] = (_Float16)ang;    // ang_in[2]
  }
  stage_fence();   // W_NRM0 + s_in ready

  // ---------------- NRM encoder (weights ping-pong: 0,1,0) ----------------
  stage_w(ws + WS_NRM1, s_w[1], 12288, tid);
  conv0_layer<1>(s_in, s_w[0], s_act[0], nrm_b0, 0, tid);
  stage_fence();

  stage_w(ws + WS_NRM2, s_w[0], 12288, tid);
  convN_layer<false>(s_act[0], s_w[1], s_act[1], nrm_b1, s_hold, tid);
  stage_fence();

  stage_w(ws + WS_ANG0, s_w[1], 2048, tid);
  convN_layer<true>(s_act[1], s_w[0], s_act[0], nrm_b2, s_hold, tid);  // nrm_e -> s_hold
  stage_fence();

  // ---------------- ANG encoder (weights ping-pong: 1,0,1) ----------------
  stage_w(ws + WS_ANG1, s_w[0], 12288, tid);
  conv0_layer<3>(s_in, s_w[1], s_act[0], ang_b0, 1, tid);
  stage_fence();

  stage_w(ws + WS_ANG2, s_w[1], 12288, tid);
  convN_layer<false>(s_act[0], s_w[0], s_act[1], ang_b1, s_hold, tid);
  stage_fence();

  stage_w(ws + WS_MLP, s_w[0], 8192, tid);
  convN_layer<false>(s_act[1], s_w[1], s_act[0], ang_b2, s_hold, tid);
  stage_fence();

  // ---------------- MLP + store ----------------
  mlp_layer(s_hold, s_act[0], s_w[0], mlp_b, out, (long)blockIdx.x * 144, tid);
}

extern "C" void kernel_launch(void* const* d_in, const int* in_sizes, int n_in,
                              void* d_out, int out_size, void* d_ws, size_t ws_size,
                              hipStream_t stream) {
  (void)in_sizes; (void)n_in; (void)out_size; (void)ws_size;
  const float* pc     = (const float*)d_in[0];
  const float* feats  = (const float*)d_in[1];
  // pos_* (2..7) and rel_* (8..13) are dead in the reference (pos_e/rel_e discarded)
  const float* nrm_w0 = (const float*)d_in[14];
  const float* nrm_b0 = (const float*)d_in[15];
  const float* nrm_w1 = (const float*)d_in[16];
  const float* nrm_b1 = (const float*)d_in[17];
  const float* nrm_w2 = (const float*)d_in[18];
  const float* nrm_b2 = (const float*)d_in[19];
  const float* ang_w0 = (const float*)d_in[20];
  const float* ang_b0 = (const float*)d_in[21];
  const float* ang_w1 = (const float*)d_in[22];
  const float* ang_b1 = (const float*)d_in[23];
  const float* ang_w2 = (const float*)d_in[24];
  const float* ang_b2 = (const float*)d_in[25];
  const float* mlp_w  = (const float*)d_in[26];
  const float* mlp_b  = (const float*)d_in[27];
  _Float16* ws = (_Float16*)d_ws;
  float* out = (float*)d_out;

  prep_weights<<<8,  256, 0, stream>>>(nrm_w0, ws + WS_NRM0, 1, 32);
  prep_weights<<<48, 256, 0, stream>>>(nrm_w1, ws + WS_NRM1, 64, 192);
  prep_weights<<<48, 256, 0, stream>>>(nrm_w2, ws + WS_NRM2, 64, 192);
  prep_weights<<<8,  256, 0, stream>>>(ang_w0, ws + WS_ANG0, 3, 32);
  prep_weights<<<48, 256, 0, stream>>>(ang_w1, ws + WS_ANG1, 64, 192);
  prep_weights<<<48, 256, 0, stream>>>(ang_w2, ws + WS_ANG2, 64, 192);
  prep_weights<<<32, 256, 0, stream>>>(mlp_w,  ws + WS_MLP,  0, 128);

  // 80000 points / 16 per block = 5000 blocks (exact)
  locse_main<<<5000, 256, 0, stream>>>(pc, ws, nrm_b0, nrm_b1, nrm_b2,
                                       ang_b0, ang_b1, ang_b2, mlp_b, out);

  // 720000*64 floats / 4 = 11,520,000 float4s = 45000 blocks * 256 (exact)
  feats_copy<<<45000, 256, 0, stream>>>((const v4f*)feats, out);
}
